// AMGSRN_8160437862447
// MI455X (gfx1250) — matrix operand order repaired and checked by reading, not compiled, not run on hardware
//
#include <hip/hip_runtime.h>
#include <hip/hip_bf16.h>

typedef __attribute__((ext_vector_type(16))) _Float16 v16h;
typedef __attribute__((ext_vector_type(8)))  _Float16 v8h;
typedef __attribute__((ext_vector_type(8)))  float    v8f;
typedef __attribute__((ext_vector_type(4)))  float    v4f;

#define N_GRIDS       64
#define GRID_RES      64
#define FEAT_DIM      128      // N_GRIDS * N_FEATS
#define NODES         128
#define PTS_PER_BLOCK 128
#define THREADS       256
#define INV_GRID_SCALING (1.0f/1.48f)
#define LRELU_SLOPE   0.2f

__device__ __forceinline__ v16h cat8(v8h lo, v8h hi) {
    union { v16h v; v8h h[2]; } u;
    u.h[0] = lo; u.h[1] = hi;
    return u.v;
}

// ---- B-fragment loaders (32x16 B matrix: lane's K values p[0..7], p[16..23] of
// its output column). Templated on stored weight type. ----
template <typename WT>
__device__ __forceinline__ v16h load_bfrag(const WT* p);

template <>
__device__ __forceinline__ v16h load_bfrag<float>(const float* p) {
    v4f a = *(const v4f*)(p);
    v4f b = *(const v4f*)(p + 4);
    v4f c = *(const v4f*)(p + 16);
    v4f d = *(const v4f*)(p + 20);
    v16h r;
#pragma unroll
    for (int j = 0; j < 4; ++j) {
        r[j]      = (_Float16)a[j];
        r[4 + j]  = (_Float16)b[j];
        r[8 + j]  = (_Float16)c[j];
        r[12 + j] = (_Float16)d[j];
    }
    return r;
}

template <>
__device__ __forceinline__ v16h load_bfrag<_Float16>(const _Float16* p) {
    v8h lo = *(const v8h*)(p);       // 16 B
    v8h hi = *(const v8h*)(p + 16);  // 16 B
    return cat8(lo, hi);
}

// One fused 128 -> 128 LReLU layer: LDS f16 activations x weights, f32
// accumulation via v_wmma_f32_16x16x32_f16. Wave owns rows 16w..16w+15.
template <typename WT>
__device__ __forceinline__ void layer128(const _Float16* __restrict__ inLDS,
                                         _Float16* __restrict__ outLDS,
                                         const WT* __restrict__ W,
                                         const float* __restrict__ b,
                                         int wave, int lane) {
    const int l16 = lane & 15;
    const int lh  = lane >> 4;   // 0: K-chunks {0..7,16..23}; 1: {8..15,24..31}

    // Preload 4 A fragments (16x32 f16) for this wave's M-tile from LDS.
    const _Float16* arow = inLDS + (wave * 16 + l16) * FEAT_DIM;
    v16h afrag[4];
#pragma unroll
    for (int k = 0; k < 4; ++k) {
        const int kb = k * 32 + lh * 8;
        v8h lo = *(const v8h*)(arow + kb);        // K = kb .. kb+7
        v8h hi = *(const v8h*)(arow + kb + 16);   // K = kb+16 .. kb+23
        afrag[k] = cat8(lo, hi);
    }

#pragma unroll
    for (int nt = 0; nt < 8; ++nt) {
        const int   col  = nt * 16 + l16;         // output node for this lane
        const float bias = b[col];
        v8f acc = { bias, bias, bias, bias, bias, bias, bias, bias };
        const WT* wrow = W + col * NODES + lh * 8;
#pragma unroll
        for (int k = 0; k < 4; ++k) {
            v16h bf = load_bfrag<WT>(wrow + k * 32);
            acc = __builtin_amdgcn_wmma_f32_16x16x32_f16(
                false, afrag[k], false, bf, (short)0, acc, false, false);
        }
        // D layout: VGPR r -> M = r + 8*lh, N = l16. Bias already folded in C.
        // leaky_relu(v, s) == max(v, s*v) for s in (0,1): v_mul + v_max_f32.
#pragma unroll
        for (int r = 0; r < 8; ++r) {
            float v = acc[r];
            v = fmaxf(v, LRELU_SLOPE * v);
            const int row = wave * 16 + lh * 8 + r;
            outLDS[row * NODES + col] = (_Float16)v;
        }
    }
}

// One-time (per launch) weight pack: W0|W1 f32 -> f16 in workspace.
__global__ __launch_bounds__(THREADS)
void pack_weights(const float* __restrict__ W0,
                  const float* __restrict__ W1,
                  _Float16* __restrict__ w16) {
    const int n = NODES * NODES;
    const int i = blockIdx.x * THREADS + threadIdx.x;
    if (i < n)          w16[i] = (_Float16)W0[i];
    else if (i < 2 * n) w16[i] = (_Float16)W1[i - n];
}

template <typename WT>
__global__ __launch_bounds__(THREADS)
void amgsrn_fused(const float* __restrict__ x,
                  const float* __restrict__ gscale,
                  const float* __restrict__ gtrans,
                  const float* __restrict__ fg,
                  const WT* __restrict__ W0, const float* __restrict__ b0,
                  const WT* __restrict__ W1, const float* __restrict__ b1,
                  const float* __restrict__ W2, const float* __restrict__ b2,
                  float* __restrict__ out) {
    __shared__ _Float16 featsLDS[PTS_PER_BLOCK * FEAT_DIM]; // 32 KB (reused as h2)
    __shared__ _Float16 hLDS[PTS_PER_BLOCK * NODES];        // 32 KB

    const int tid       = threadIdx.x;
    const int blockBase = blockIdx.x * PTS_PER_BLOCK;

    // ---------- Phase 1: 64-grid trilinear sampling -> LDS f16 ----------
    {
        const int p  = tid >> 1;          // point 0..127 in block
        const int g0 = (tid & 1) * 32;    // each thread samples 32 grids
        const float px = x[(blockBase + p) * 3 + 0];
        const float py = x[(blockBase + p) * 3 + 1];
        const float pz = x[(blockBase + p) * 3 + 2];

        for (int gi = 0; gi < 32; ++gi) {
            const int g = g0 + gi;
            const float cx = (px * gscale[g * 3 + 0] + gtrans[g * 3 + 0]) * INV_GRID_SCALING;
            const float cy = (py * gscale[g * 3 + 1] + gtrans[g * 3 + 1]) * INV_GRID_SCALING;
            const float cz = (pz * gscale[g * 3 + 2] + gtrans[g * 3 + 2]) * INV_GRID_SCALING;
            // align_corners=True: f = (c+1)*0.5*(R-1); x->W, y->H, z->D
            const float fx = (cx + 1.0f) * (0.5f * (GRID_RES - 1));
            const float fy = (cy + 1.0f) * (0.5f * (GRID_RES - 1));
            const float fz = (cz + 1.0f) * (0.5f * (GRID_RES - 1));
            const float x0f = floorf(fx), y0f = floorf(fy), z0f = floorf(fz);
            const float wx = fx - x0f, wy = fy - y0f, wz = fz - z0f;
            const int ix0 = (int)x0f, iy0 = (int)y0f, iz0 = (int)z0f;
            const int ix1 = ix0 + 1, iy1 = iy0 + 1, iz1 = iz0 + 1;

            int   idx[8];
            float msk[8];
#pragma unroll
            for (int c = 0; c < 8; ++c) {
                const int xi = (c & 1) ? ix1 : ix0;
                const int yi = (c & 2) ? iy1 : iy0;
                const int zi = (c & 4) ? iz1 : iz0;
                const bool ok = ((unsigned)xi < (unsigned)GRID_RES) &&
                                ((unsigned)yi < (unsigned)GRID_RES) &&
                                ((unsigned)zi < (unsigned)GRID_RES);
                const int xc = min(max(xi, 0), GRID_RES - 1);
                const int yc = min(max(yi, 0), GRID_RES - 1);
                const int zc = min(max(zi, 0), GRID_RES - 1);
                idx[c] = (zc << 12) | (yc << 6) | xc;
                msk[c] = ok ? 1.0f : 0.0f;
            }
            const float* g0p = fg + (size_t)g * (size_t)(2 * GRID_RES * GRID_RES * GRID_RES);
            const float* g1p = g0p + (GRID_RES * GRID_RES * GRID_RES);

            float v0[8], v1[8];
#pragma unroll
            for (int c = 0; c < 8; ++c) {
                v0[c] = g0p[idx[c]] * msk[c];
                v1[c] = g1p[idx[c]] * msk[c];
            }
            auto tri = [&](const float* v) -> float {
                const float c00 = v[0] + (v[1] - v[0]) * wx;
                const float c01 = v[2] + (v[3] - v[2]) * wx;
                const float c10 = v[4] + (v[5] - v[4]) * wx;
                const float c11 = v[6] + (v[7] - v[6]) * wx;
                const float c0  = c00 + (c01 - c00) * wy;
                const float c1  = c10 + (c11 - c10) * wy;
                return c0 + (c1 - c0) * wz;
            };
            const _Float16 h0 = (_Float16)tri(v0);
            const _Float16 h1 = (_Float16)tri(v1);
            const unsigned pk =
                (unsigned)__builtin_bit_cast(unsigned short, h0) |
                ((unsigned)__builtin_bit_cast(unsigned short, h1) << 16);
            *(unsigned*)&featsLDS[p * FEAT_DIM + 2 * g] = pk;  // [pt][2g..2g+1]
        }
    }
    __syncthreads();

    // ---------- Phase 2: decoder via WMMA ----------
    const int wave = tid >> 5;
    const int lane = tid & 31;
    layer128<WT>(featsLDS, hLDS, W0, b0, wave, lane);   // feats -> h1
    layer128<WT>(hLDS, featsLDS, W1, b1, wave, lane);   // h1 -> h2 (reuse buffer)
    __syncthreads();

    // ---------- Phase 3: 128 -> 1 linear (vectorized dot) ----------
    if (tid < PTS_PER_BLOCK) {
        const _Float16* hrow = featsLDS + tid * NODES;
        float acc = b2[0];
#pragma unroll
        for (int k0 = 0; k0 < NODES; k0 += 8) {
            v8h hv  = *(const v8h*)(hrow + k0);     // one ds_load_b128
            v4f wa  = *(const v4f*)(W2 + k0);
            v4f wb  = *(const v4f*)(W2 + k0 + 4);
            acc += (float)hv[0] * wa[0] + (float)hv[1] * wa[1] +
                   (float)hv[2] * wa[2] + (float)hv[3] * wa[3] +
                   (float)hv[4] * wb[0] + (float)hv[5] * wb[1] +
                   (float)hv[6] * wb[2] + (float)hv[7] * wb[3];
        }
        out[blockBase + tid] = acc;
    }
}

extern "C" void kernel_launch(void* const* d_in, const int* in_sizes, int n_in,
                              void* d_out, int out_size, void* d_ws, size_t ws_size,
                              hipStream_t stream) {
    const float* x  = (const float*)d_in[0];
    const float* gs = (const float*)d_in[1];
    const float* gt = (const float*)d_in[2];
    const float* fg = (const float*)d_in[3];
    const float* W0 = (const float*)d_in[4];
    const float* b0 = (const float*)d_in[5];
    const float* W1 = (const float*)d_in[6];
    const float* b1 = (const float*)d_in[7];
    const float* W2 = (const float*)d_in[8];
    const float* b2 = (const float*)d_in[9];
    float* out = (float*)d_out;

    const int npts   = in_sizes[0] / 3;             // 1048576
    const int blocks = npts / PTS_PER_BLOCK;        // 8192

    const size_t packedBytes = (size_t)(2 * NODES * NODES) * sizeof(_Float16); // 64 KB
    if (ws_size >= packedBytes) {
        // Fast path: pack weights to f16 once per launch, then fused kernel
        // loads B-fragments directly as f16 (no per-wave cvt).
        _Float16* w16 = (_Float16*)d_ws;
        pack_weights<<<(2 * NODES * NODES + THREADS - 1) / THREADS, THREADS, 0, stream>>>(
            W0, W1, w16);
        amgsrn_fused<_Float16><<<blocks, THREADS, 0, stream>>>(
            x, gs, gt, fg, w16, b0, w16 + NODES * NODES, b1, W2, b2, out);
    } else {
        // Fallback: convert f32 weights to f16 on the fly inside the kernel.
        amgsrn_fused<float><<<blocks, THREADS, 0, stream>>>(
            x, gs, gt, fg, W0, b0, W1, b1, W2, b2, out);
    }
}
